// MultiHeadAttention_5093831213374
// MI455X (gfx1250) — matrix operand order repaired and checked by reading, not compiled, not run on hardware
//
#include <hip/hip_runtime.h>

#define DMODEL 2048
#define SEQ    2048
#define BATCH  2
#define HEADS  16
#define DK     128

// padded LDS row strides (elements) to spread rows across banks
#define SAW 40    // GEMM A panel row stride (32 data + 8 pad)
#define SBW 40    // GEMM B panel row stride
#define SKW 136   // attention K panel row stride (128 + 8)
#define SVW 72    // attention V panel row stride (64 + 8)
#define SPW 72    // attention P bounce row stride

typedef __bf16 bf16;
typedef bf16  bf16x16 __attribute__((ext_vector_type(16)));
typedef float f32x8   __attribute__((ext_vector_type(8)));
typedef int   i32x4   __attribute__((ext_vector_type(4)));

typedef __attribute__((address_space(1))) i32x4* gvec_ptr;  // global <4 x i32>*
typedef __attribute__((address_space(3))) i32x4* lvec_ptr;  // LDS    <4 x i32>*

#if __has_builtin(__builtin_amdgcn_global_load_async_to_lds_b128) && \
    __has_builtin(__builtin_amdgcn_s_wait_asynccnt)
#define HAVE_ASYNC 1
#else
#define HAVE_ASYNC 0
#endif

// ---------------------------------------------------------------------------
// WMMA wrapper: D(f32 16x16) = A(bf16 16x32) * B(bf16 32x16) + C
// ---------------------------------------------------------------------------
__device__ __forceinline__ f32x8 wmma_bf16(bf16x16 a, bf16x16 b, f32x8 c) {
    return __builtin_amdgcn_wmma_f32_16x16x32_bf16(
        false, a, false, b, (short)0, c, false, false);
}

union FragCast { i32x4 i[2]; bf16x16 v; };

// A fragment (16x32): lane L (row=L&15, g=L>>4) holds K in [8g,8g+8) U [16+8g,16+8g+8)
__device__ __forceinline__ bf16x16 frag_a(const bf16* t, int ld, int lane) {
    int row = lane & 15, g = lane >> 4;
    const bf16* p = t + (size_t)row * ld + 8 * g;
    FragCast u;
    u.i[0] = *(const i32x4*)p;
    u.i[1] = *(const i32x4*)(p + 16);
    return u.v;
}

// B fragment (32x16), source = B^T row-major (row = out col, contiguous K):
// lane L (col=L&15, g=L>>4) holds K in [8g,8g+8) U [16+8g,16+8g+8)
__device__ __forceinline__ bf16x16 frag_b(const bf16* t, size_t ld, int lane) {
    int col = lane & 15, g = lane >> 4;
    const bf16* p = t + (size_t)col * ld + 8 * g;
    FragCast u;
    u.i[0] = *(const i32x4*)p;
    u.i[1] = *(const i32x4*)(p + 16);
    return u.v;
}

// ---------------------------------------------------------------------------
// 16-byte global -> LDS transfer: CDNA5 async op when available
// ---------------------------------------------------------------------------
__device__ __forceinline__ void async_ld16(const bf16* g, bf16* l) {
#if HAVE_ASYNC
    __builtin_amdgcn_global_load_async_to_lds_b128(
        (gvec_ptr)(i32x4*)(void*)g,
        (lvec_ptr)(i32x4*)(void*)l, 0, 0);
#else
    *(i32x4*)l = *(const i32x4*)g;
#endif
}

template <int N>
__device__ __forceinline__ void wait_async() {
#if HAVE_ASYNC
    __builtin_amdgcn_s_wait_asynccnt(N);
#endif
}

// ---------------------------------------------------------------------------
// GEMM mainloop: 256 threads (8 waves, 4(M)x2(N)), block tile 256x128,
// wave tile 64x64 (4x4 WMMA tiles), K step 32, double-buffered async LDS.
// ---------------------------------------------------------------------------
template <int KDIM>
__device__ __forceinline__ void gemm_mainloop(const bf16* __restrict__ A,
                                              const bf16* __restrict__ Bt,
                                              int m0, int n0,
                                              bf16* sA, bf16* sB,
                                              f32x8 acc[4][4]) {
    const int tid  = threadIdx.x;
    const int lane = tid & 31;
    const int wave = tid >> 5;
    const int wm   = wave >> 1;   // 0..3
    const int wn   = wave & 1;    // 0..1
    const int brow = tid & 127;
    const int bhh  = tid >> 7;

#pragma unroll
    for (int i = 0; i < 4; i++)
#pragma unroll
        for (int j = 0; j < 4; j++)
#pragma unroll
            for (int e = 0; e < 8; e++) acc[i][j][e] = 0.f;

    auto issue = [&](int buf, int kt) {
        // A panel: 256 rows x 32 K, one row (64B = 4x16B) per thread
        const bf16* gA = A + (size_t)(m0 + tid) * KDIM + kt;
        bf16* lA = sA + buf * (256 * SAW) + tid * SAW;
#pragma unroll
        for (int c = 0; c < 4; c++) async_ld16(gA + 8 * c, lA + 8 * c);
        // B panel: 128 rows x 32 K, half row (32B = 2x16B) per thread
        const bf16* gB = Bt + (size_t)(n0 + brow) * KDIM + kt + 16 * bhh;
        bf16* lB = sB + buf * (128 * SBW) + brow * SBW + 16 * bhh;
#pragma unroll
        for (int c = 0; c < 2; c++) async_ld16(gB + 8 * c, lB + 8 * c);
    };

    issue(0, 0);
    int buf = 0;
    for (int kt = 0; kt < KDIM; kt += 32, buf ^= 1) {
        if (kt + 32 < KDIM) { issue(buf ^ 1, kt + 32); wait_async<6>(); }
        else                { wait_async<0>(); }
        __syncthreads();
        const bf16* cA = sA + buf * (256 * SAW);
        const bf16* cB = sB + buf * (128 * SBW);
        bf16x16 af[4], bfr[4];
#pragma unroll
        for (int i = 0; i < 4; i++) af[i]  = frag_a(cA + (wm * 64 + i * 16) * SAW, SAW, lane);
#pragma unroll
        for (int j = 0; j < 4; j++) bfr[j] = frag_b(cB + (wn * 64 + j * 16) * SBW, SBW, lane);
#pragma unroll
        for (int i = 0; i < 4; i++)
#pragma unroll
            for (int j = 0; j < 4; j++)
                acc[i][j] = wmma_bf16(af[i], bfr[j], acc[i][j]);
        __syncthreads();
    }
}

// ---------------------------------------------------------------------------
// Prep kernels
// ---------------------------------------------------------------------------
__global__ void cvt_x_bf16_kernel(const float* __restrict__ src, bf16* __restrict__ dst) {
    size_t i = ((size_t)blockIdx.x * blockDim.x + threadIdx.x) * 8;
#pragma unroll
    for (int e = 0; e < 8; e++) dst[i + e] = (bf16)src[i + e];
}

__global__ void transpose_w_bf16_kernel(const float* __restrict__ src, bf16* __restrict__ dst) {
    __shared__ float t[32][33];
    int k0 = blockIdx.x * 32, n0 = blockIdx.y * 32;
    for (int i = threadIdx.y; i < 32; i += 8)
        t[i][threadIdx.x] = src[(size_t)(k0 + i) * DMODEL + n0 + threadIdx.x];
    __syncthreads();
    for (int i = threadIdx.y; i < 32; i += 8)
        dst[(size_t)(n0 + i) * DMODEL + k0 + threadIdx.x] = (bf16)t[threadIdx.x][i];
}

__global__ void rope_tab_kernel(float* __restrict__ cosT, float* __restrict__ sinT) {
    int t = blockIdx.x * blockDim.x + threadIdx.x;
    int pos = t >> 6, j = t & 63;
    float inv = __expf(-(float)(2 * j) * (9.210340371976184f / 128.f));
    float ang = (float)pos * inv;
    cosT[t] = cosf(ang);
    sinT[t] = sinf(ang);
}

// ---------------------------------------------------------------------------
// Fused QKV projection + RoPE + head-major scatter (V stored transposed)
// ---------------------------------------------------------------------------
__global__ void __launch_bounds__(256)
gemm_qkv_rope_kernel(const bf16* __restrict__ Xb, const bf16* __restrict__ Wt,
                     const int* __restrict__ tp,
                     const float* __restrict__ cosT, const float* __restrict__ sinT,
                     bf16* __restrict__ Qr, bf16* __restrict__ Kr, bf16* __restrict__ Vt) {
    __shared__ __align__(16) bf16 sA[2 * 256 * SAW];
    __shared__ __align__(16) bf16 sB[2 * 128 * SBW];
    f32x8 acc[4][4];
    const int m0 = blockIdx.x * 256, n0 = blockIdx.y * 128;
    gemm_mainloop<DMODEL>(Xb, Wt, m0, n0, sA, sB, acc);

    const int lane = threadIdx.x & 31;
    const int wave = threadIdx.x >> 5;
    const int wm = wave >> 1, wn = wave & 1;
    const int hg = lane >> 4, ln = lane & 15;
#pragma unroll
    for (int i = 0; i < 4; i++) {
#pragma unroll
        for (int j = 0; j < 4; j++) {
            int N     = n0 + wn * 64 + j * 16 + ln;
            int which = N >> 11;         // 0=Q 1=K 2=V (uniform per tile)
            int nn    = N & 2047;
            int hh    = nn >> 7;
            int d     = nn & 127;
            int jj    = d >> 1;
            int mb    = m0 + wm * 64 + i * 16 + (hg << 3);
#pragma unroll
            for (int r = 0; r < 8; r++) {
                int M = mb + r;
                float v = acc[i][j][r];
                float partner = __shfl_xor(v, 1, 32);
                int b = M >> 11, s = M & 2047;
                if (which == 2) {
                    Vt[(((size_t)(b * HEADS + hh)) * DK + d) * SEQ + s] = (bf16)v;
                } else {
                    int pos = tp[M];
                    float cs = cosT[pos * 64 + jj];
                    float sn = sinT[pos * 64 + jj];
                    float out = (d & 1) ? (v * cs + partner * sn)
                                        : (v * cs - partner * sn);
                    bf16* dst = (which == 0) ? Qr : Kr;
                    dst[(((size_t)(b * HEADS + hh)) * SEQ + s) * DK + d] = (bf16)out;
                }
            }
        }
    }
}

// ---------------------------------------------------------------------------
// Flash attention: 4 waves/block, 16 query rows per wave, 64-key blocks.
// K/V tiles double-buffered in LDS via async loads (uniform trip count
// blockIdx.x+1 across all waves, so block-wide barriers are legal).
// ---------------------------------------------------------------------------
__global__ void __launch_bounds__(128)
attn_kernel(const bf16* __restrict__ Qr, const bf16* __restrict__ Kr,
            const bf16* __restrict__ Vt, bf16* __restrict__ Ob) {
    __shared__ __align__(16) bf16 sK[2 * 64 * SKW];
    __shared__ __align__(16) bf16 sV[2 * 128 * SVW];
    __shared__ __align__(16) bf16 pP[4 * 16 * SPW];
    const int bh = blockIdx.y;
    const int b = bh >> 4, h = bh & 15;
    const int tid = threadIdx.x;
    const int wave = tid >> 5, lane = tid & 31;
    const int q0 = blockIdx.x * 64 + wave * 16;
    const bf16* Qh = Qr + (size_t)bh * SEQ * DK;
    const bf16* Kh = Kr + (size_t)bh * SEQ * DK;
    const bf16* Vh = Vt + (size_t)bh * DK * SEQ;
    bf16* myP = pP + wave * (16 * SPW);

    bf16x16 qa[4];
#pragma unroll
    for (int c = 0; c < 4; c++) qa[c] = frag_a(Qh + (size_t)q0 * DK + 32 * c, DK, lane);

    float m[8], l[8];
    f32x8 o[8];
#pragma unroll
    for (int r = 0; r < 8; r++) { m[r] = -1e30f; l[r] = 0.f; }
#pragma unroll
    for (int dn = 0; dn < 8; dn++)
#pragma unroll
        for (int e = 0; e < 8; e++) o[dn][e] = 0.f;

    const float scale = 0.08838834764831845f;  // 1/sqrt(128)
    const int hg = lane >> 4;
    const int ln = lane & 15;

    auto issueKV = [&](int buf, int kb) {
        // K tile: 64 keys x 128 d ; half row (128B = 8x16B) per thread
        const bf16* gK = Kh + (size_t)(kb + (tid >> 1)) * DK + (tid & 1) * 64;
        bf16* lK = sK + buf * (64 * SKW) + (tid >> 1) * SKW + (tid & 1) * 64;
#pragma unroll
        for (int c = 0; c < 8; c++) async_ld16(gK + 8 * c, lK + 8 * c);
        // V tile: 128 d x 64 keys ; one row (128B = 8x16B) per thread
        const bf16* gV = Vh + (size_t)tid * SEQ + kb;
        bf16* lV = sV + buf * (128 * SVW) + tid * SVW;
#pragma unroll
        for (int c = 0; c < 8; c++) async_ld16(gV + 8 * c, lV + 8 * c);
    };

    const int nblk = blockIdx.x + 1;   // uniform for all waves in block
    issueKV(0, 0);
    int buf = 0;
    for (int ib = 0; ib < nblk; ib++, buf ^= 1) {
        const int kb = ib * 64;
        if (ib + 1 < nblk) { issueKV(buf ^ 1, kb + 64); wait_async<16>(); }
        else               { wait_async<0>(); }
        __syncthreads();
        const bf16* cK = sK + buf * (64 * SKW);
        const bf16* cV = sV + buf * (128 * SVW);

        // ---- scores S = Q K^T (16 queries x 64 keys) ----
        f32x8 sc[4];
#pragma unroll
        for (int nt = 0; nt < 4; nt++) {
#pragma unroll
            for (int e = 0; e < 8; e++) sc[nt][e] = 0.f;
#pragma unroll
            for (int c = 0; c < 4; c++) {
                bf16x16 kf = frag_b(cK + (nt * 16) * SKW + 32 * c, SKW, lane);
                sc[nt] = wmma_bf16(qa[c], kf, sc[nt]);
            }
        }
        // ---- online softmax ----
        float pexp[4][8];
        float alpha[8];
#pragma unroll
        for (int r = 0; r < 8; r++) {
            int query = q0 + r + 8 * hg;
            float sv[4];
            float vmax = -1e30f;
#pragma unroll
            for (int nt = 0; nt < 4; nt++) {
                int key = kb + nt * 16 + ln;
                float s = sc[nt][r] * scale;
                s = (key > query) ? -1e30f : s;
                sv[nt] = s;
                vmax = fmaxf(vmax, s);
            }
#pragma unroll
            for (int off = 1; off < 16; off <<= 1) vmax = fmaxf(vmax, __shfl_xor(vmax, off, 32));
            float mn = fmaxf(m[r], vmax);
            float al = __expf(m[r] - mn);
            float rs = 0.f;
#pragma unroll
            for (int nt = 0; nt < 4; nt++) {
                float p = __expf(sv[nt] - mn);
                pexp[nt][r] = p;
                rs += p;
            }
#pragma unroll
            for (int off = 1; off < 16; off <<= 1) rs += __shfl_xor(rs, off, 32);
            l[r] = l[r] * al + rs;
            m[r] = mn;
            alpha[r] = al;
        }
#pragma unroll
        for (int dn = 0; dn < 8; dn++)
#pragma unroll
            for (int r = 0; r < 8; r++) o[dn][r] *= alpha[r];
        // ---- bounce P through LDS: C layout -> A layout ----
#pragma unroll
        for (int nt = 0; nt < 4; nt++)
#pragma unroll
            for (int r = 0; r < 8; r++) {
                int row = r + 8 * hg;
                myP[row * SPW + nt * 16 + ln] = (bf16)pexp[nt][r];
            }
        bf16x16 pa[2];
#pragma unroll
        for (int c = 0; c < 2; c++) pa[c] = frag_a(myP + 32 * c, SPW, lane);
        // ---- O += P V ----
#pragma unroll
        for (int dn = 0; dn < 8; dn++) {
#pragma unroll
            for (int c = 0; c < 2; c++) {
                bf16x16 vf = frag_b(cV + (dn * 16) * SVW + 32 * c, SVW, lane);
                o[dn] = wmma_bf16(pa[c], vf, o[dn]);
            }
        }
        __syncthreads();
    }
    // ---- normalize + write O (bf16, [B,S,H*DK]) ----
#pragma unroll
    for (int r = 0; r < 8; r++) l[r] = 1.f / l[r];
#pragma unroll
    for (int dn = 0; dn < 8; dn++)
#pragma unroll
        for (int r = 0; r < 8; r++) {
            int query = q0 + r + 8 * hg;
            int col = h * DK + dn * 16 + ln;
            Ob[((size_t)(b * SEQ + query)) * DMODEL + col] = (bf16)(o[dn][r] * l[r]);
        }
}

// ---------------------------------------------------------------------------
// Output projection: out(f32) = Ob(bf16) @ Wo
// ---------------------------------------------------------------------------
__global__ void __launch_bounds__(256)
gemm_out_kernel(const bf16* __restrict__ Ob, const bf16* __restrict__ Wot,
                float* __restrict__ out) {
    __shared__ __align__(16) bf16 sA[2 * 256 * SAW];
    __shared__ __align__(16) bf16 sB[2 * 128 * SBW];
    f32x8 acc[4][4];
    const int m0 = blockIdx.x * 256, n0 = blockIdx.y * 128;
    gemm_mainloop<DMODEL>(Ob, Wot, m0, n0, sA, sB, acc);

    const int lane = threadIdx.x & 31;
    const int wave = threadIdx.x >> 5;
    const int wm = wave >> 1, wn = wave & 1;
    const int hg = lane >> 4, ln = lane & 15;
#pragma unroll
    for (int i = 0; i < 4; i++) {
#pragma unroll
        for (int j = 0; j < 4; j++) {
            int N  = n0 + wn * 64 + j * 16 + ln;
            int mb = m0 + wm * 64 + i * 16 + (hg << 3);
#pragma unroll
            for (int r = 0; r < 8; r++)
                out[(size_t)(mb + r) * DMODEL + N] = acc[i][j][r];
        }
    }
}

// ---------------------------------------------------------------------------
extern "C" void kernel_launch(void* const* d_in, const int* in_sizes, int n_in,
                              void* d_out, int out_size, void* d_ws, size_t ws_size,
                              hipStream_t stream) {
    (void)in_sizes; (void)n_in; (void)out_size; (void)ws_size;
    const float* x  = (const float*)d_in[0];
    const int*   tp = (const int*)d_in[1];
    const float* wq = (const float*)d_in[2];
    const float* wk = (const float*)d_in[3];
    const float* wv = (const float*)d_in[4];
    const float* wo = (const float*)d_in[5];
    float* out = (float*)d_out;

    char* ws = (char*)d_ws;
    size_t off = 0;
    auto take = [&](size_t bytes) -> char* {
        char* p = ws + off;
        off = (off + bytes + 255) & ~(size_t)255;
        return p;
    };
    const size_t MD = (size_t)BATCH * SEQ * DMODEL;
    bf16* Xb   = (bf16*)take(MD * sizeof(bf16));
    bf16* Wqkv = (bf16*)take((size_t)3 * DMODEL * DMODEL * sizeof(bf16));
    bf16* Wot  = (bf16*)take((size_t)DMODEL * DMODEL * sizeof(bf16));
    bf16* Qr   = (bf16*)take(MD * sizeof(bf16));
    bf16* Kr   = (bf16*)take(MD * sizeof(bf16));
    bf16* Vt   = (bf16*)take(MD * sizeof(bf16));
    bf16* Ob   = (bf16*)take(MD * sizeof(bf16));
    float* cosT = (float*)take((size_t)SEQ * 64 * sizeof(float));
    float* sinT = (float*)take((size_t)SEQ * 64 * sizeof(float));

    cvt_x_bf16_kernel<<<(int)(MD / (256 * 8)), 256, 0, stream>>>(x, Xb);
    transpose_w_bf16_kernel<<<dim3(64, 64), dim3(32, 8), 0, stream>>>(wq, Wqkv);
    transpose_w_bf16_kernel<<<dim3(64, 64), dim3(32, 8), 0, stream>>>(wk, Wqkv + (size_t)DMODEL * DMODEL);
    transpose_w_bf16_kernel<<<dim3(64, 64), dim3(32, 8), 0, stream>>>(wv, Wqkv + (size_t)2 * DMODEL * DMODEL);
    transpose_w_bf16_kernel<<<dim3(64, 64), dim3(32, 8), 0, stream>>>(wo, Wot);
    rope_tab_kernel<<<(SEQ * 64) / 256, 256, 0, stream>>>(cosT, sinT);

    gemm_qkv_rope_kernel<<<dim3(16, 48), 256, 0, stream>>>(Xb, Wqkv, tp, cosT, sinT, Qr, Kr, Vt);
    attn_kernel<<<dim3(SEQ / 64, BATCH * HEADS), 128, 0, stream>>>(Qr, Kr, Vt, Ob);
    gemm_out_kernel<<<dim3(16, 16), 256, 0, stream>>>(Ob, Wot, out);
}
